// GCN_90494960926917
// MI455X (gfx1250) — matrix operand order repaired and checked by reading, not compiled, not run on hardware
//
#include <hip/hip_runtime.h>

#define NEG_SLOPE 0.01f

typedef __attribute__((ext_vector_type(16))) _Float16 v16h;
typedef __attribute__((ext_vector_type(8)))  float    v8f;

// ---------------- utility kernels ----------------
__global__ void k_zero(float* __restrict__ p, int n) {
  int t = blockIdx.x * blockDim.x + threadIdx.x;
  if (t < n) p[t] = 0.0f;
}

__global__ void k_degrees(const int* __restrict__ src, const int* __restrict__ dst,
                          float* __restrict__ dout, float* __restrict__ din, int nE) {
  int e = blockIdx.x * blockDim.x + threadIdx.x;
  if (e >= nE) return;
  atomicAdd(&dout[src[e]], 1.0f);
  atomicAdd(&din[dst[e]], 1.0f);
}

__global__ void k_invsqrt(float* __restrict__ d, int n) {
  int t = blockIdx.x * blockDim.x + threadIdx.x;
  if (t >= n) return;
  float v = d[t];
  v = v < 1.0f ? 1.0f : v;      // clip(deg, 1, None)
  d[t] = rsqrtf(v);             // deg^{-1/2}
}

// ---------------- fused normalize + dense transform via WMMA ----------------
// T[row,:] = (X[row,:] * cs[row]) @ W      X:[N,64] f32, W:[64,64] f32 row-major
// One wave per 16-row block; W preloaded as 8 v16h B-fragments per wave.
__global__ void __launch_bounds__(128)
k_gemm_norm(const float* __restrict__ X, const float* __restrict__ cs,
            const float* __restrict__ W, float* __restrict__ T, int nRowBlocks) {
  const int lane = threadIdx.x & 31;
  const int wave = threadIdx.x >> 5;
  const int n  = lane & 15;   // column within 16-wide tile (also row-in-block for A)
  const int hi = lane >> 4;   // lane-half select

  // B fragments: 32x16 f16 tile. Layout matches A: lanes 0-15 halves 0..7 -> K=0..7, 8..15 -> K=16..23;
  // lanes 16-31 -> K=8..15 / K=24..31.  bfrag[nt][kt] covers K = kt*32 .. kt*32+31, N = nt*16 .. +15.
  v16h bfrag[4][2];
  #pragma unroll
  for (int nt = 0; nt < 4; ++nt)
    #pragma unroll
    for (int kt = 0; kt < 2; ++kt)
      #pragma unroll
      for (int j = 0; j < 16; ++j)
        bfrag[nt][kt][j] = (_Float16)W[(kt * 32 + (j >> 3) * 16 + hi * 8 + (j & 7)) * 64 + nt * 16 + n];

  const int nw = gridDim.x * 4;
  for (int rb = blockIdx.x * 4 + wave; rb < nRowBlocks; rb += nw) {
    // speculative prefetch of a future row block (safe if OOB)
    __builtin_prefetch(X + (size_t)(rb + nw) * 16 * 64, 0, 0);

    const int row = rb * 16 + n;
    const float c = cs[row];
    const float* xr = X + (size_t)row * 64;

    // A fragments: 16x32 f16 tile. ISA layout: lanes 0-15 halves 0..7 -> K=0..7,
    // halves 8..15 -> K=16..23; lanes 16-31 -> K=8..15 / K=24..31.
    v16h afrag[2];
    #pragma unroll
    for (int kt = 0; kt < 2; ++kt) {
      #pragma unroll
      for (int j = 0; j < 8; ++j) {
        afrag[kt][j]     = (_Float16)(c * xr[kt * 32 + hi * 8 + j]);
        afrag[kt][j + 8] = (_Float16)(c * xr[kt * 32 + 16 + hi * 8 + j]);
      }
    }

    #pragma unroll
    for (int nt = 0; nt < 4; ++nt) {
      v8f acc = {};
      acc = __builtin_amdgcn_wmma_f32_16x16x32_f16(false, afrag[0], false, bfrag[nt][0],
                                                   (short)0, acc, false, false);
      acc = __builtin_amdgcn_wmma_f32_16x16x32_f16(false, afrag[1], false, bfrag[nt][1],
                                                   (short)0, acc, false, false);
      // C/D layout: VGPR r -> row M = r + hi*8, column = n
      float* tr = T + (size_t)(rb * 16 + hi * 8) * 64 + nt * 16 + n;
      #pragma unroll
      for (int r = 0; r < 8; ++r) tr[(size_t)r * 64] = acc[r];
    }
  }
}

// ---------------- SpMM scatter: agg[dst,:] += T[src,:] ----------------
// 64 lanes per edge -> one fully-coalesced 256B row per edge, f32 atomics into L2.
__global__ void __launch_bounds__(256)
k_scatter(const float* __restrict__ T, const int* __restrict__ src,
          const int* __restrict__ dst, float* __restrict__ agg, int nE) {
  const int f = threadIdx.x & 63;
  const int e = blockIdx.x * 4 + (threadIdx.x >> 6);
  if (e >= nE) return;
  const int s = src[e], d = dst[e];
  atomicAdd(&agg[(size_t)d * 64 + f], T[(size_t)s * 64 + f]);
}

// ---------------- out = leaky_relu(agg * c_dst + b) ----------------
__global__ void k_finalize(const float* __restrict__ agg, const float* __restrict__ cd,
                           const float* __restrict__ b, float* __restrict__ out, int n) {
  int t = blockIdx.x * blockDim.x + threadIdx.x;
  if (t >= n) return;
  const int node = t >> 6, f = t & 63;
  float v = agg[t] * cd[node] + b[f];
  out[t] = v > 0.0f ? v : NEG_SLOPE * v;
}

// ---------------- head: out = H @ W3 + b3   (64 -> 10, exact f32) ----------------
__global__ void k_head(const float* __restrict__ H, const float* __restrict__ W3,
                       const float* __restrict__ b3, float* __restrict__ out, int nNodes) {
  int t = blockIdx.x * blockDim.x + threadIdx.x;
  int node = t / 10, c = t % 10;
  if (node >= nNodes) return;
  const float* h = H + (size_t)node * 64;
  float s = b3[c];
  #pragma unroll
  for (int k = 0; k < 64; ++k) s += h[k] * W3[k * 10 + c];
  out[(size_t)node * 10 + c] = s;
}

extern "C" void kernel_launch(void* const* d_in, const int* in_sizes, int n_in,
                              void* d_out, int out_size, void* d_ws, size_t ws_size,
                              hipStream_t stream) {
  const float* x  = (const float*)d_in[0];
  const int* esrc = (const int*)d_in[1];
  const int* edst = (const int*)d_in[2];
  const float* W1 = (const float*)d_in[3];
  const float* b1 = (const float*)d_in[4];
  const float* W2 = (const float*)d_in[5];
  const float* b2 = (const float*)d_in[6];
  const float* W3 = (const float*)d_in[7];
  const float* b3 = (const float*)d_in[8];
  float* out = (float*)d_out;

  const int N  = in_sizes[0] / 64;   // 50000
  const int E  = in_sizes[1];        // 800000
  const int NF = N * 64;
  const int rb = N / 16;             // N divisible by 16 for this workload

  // workspace layout (floats): [c_src|c_dst (2N)] [tbuf NF] [aggbuf NF] [hbuf NF]
  float* ws     = (float*)d_ws;
  float* cs     = ws;
  float* cd     = ws + N;
  float* tbuf   = ws + 2 * (size_t)N;
  float* aggbuf = tbuf + NF;
  float* hbuf   = aggbuf + NF;

  // degree normalizations
  k_zero   <<<(2 * N + 255) / 256, 256, 0, stream>>>(cs, 2 * N);
  k_degrees<<<(E + 255) / 256,     256, 0, stream>>>(esrc, edst, cs, cd, E);
  k_invsqrt<<<(2 * N + 255) / 256, 256, 0, stream>>>(cs, 2 * N);

  // ---- layer 1 ----
  k_gemm_norm<<<512, 128, 0, stream>>>(x, cs, W1, tbuf, rb);
  k_zero     <<<(NF + 255) / 256, 256, 0, stream>>>(aggbuf, NF);
  k_scatter  <<<(E + 3) / 4,      256, 0, stream>>>(tbuf, esrc, edst, aggbuf, E);
  k_finalize <<<(NF + 255) / 256, 256, 0, stream>>>(aggbuf, cd, b1, hbuf, NF);

  // ---- layer 2 ----
  k_gemm_norm<<<512, 128, 0, stream>>>(hbuf, cs, W2, tbuf, rb);
  k_zero     <<<(NF + 255) / 256, 256, 0, stream>>>(aggbuf, NF);
  k_scatter  <<<(E + 3) / 4,      256, 0, stream>>>(tbuf, esrc, edst, aggbuf, E);
  k_finalize <<<(NF + 255) / 256, 256, 0, stream>>>(aggbuf, cd, b2, hbuf, NF);

  // ---- head ----
  k_head<<<(N * 10 + 255) / 256, 256, 0, stream>>>(hbuf, W3, b3, out, N);
}
